// Normalized_Correlation_Layer_39470749450727
// MI455X (gfx1250) — matrix operand order repaired and checked by reading, not compiled, not run on hardware
//
#include <hip/hip_runtime.h>
#include <stdint.h>

typedef __attribute__((ext_vector_type(16))) _Float16 v16h;
typedef __attribute__((ext_vector_type(8)))  float    v8f;

// ---- problem constants -----------------------------------------------------
constexpr int Bn = 8, Hh = 64, Wd = 32, Cc = 64;
constexpr int PATCH = 5;
constexpr int P2W = 28;               // dense patch grid: 60 x 28
constexpr int P1W = 14;               // strided patch grid: 30 x 14
constexpr int M_TOT = 60 * 28;        // 1680 patches from input_2
constexpr int N_TOT = 30 * 14;        // 420 patches from input_1
constexpr int K_TOT = PATCH * PATCH * Cc; // 1600

// ---- GEMM tiling -----------------------------------------------------------
constexpr int BM = 128, BN = 64, BK = 32;
constexpr int LDA = 40;               // halves per LDS row (32 + 8 pad)
constexpr int NIT = K_TOT / BK;       // 50 K-steps

union AFrag { v16h v; uint4 q[2]; };

// CDNA5 async copy: memory -> LDS with no VGPR staging (tracked by ASYNCcnt).
__device__ __forceinline__ void async_load_b128(uint32_t lds_off, const void* gptr) {
    asm volatile("global_load_async_to_lds_b128 %0, %1, off"
                 :: "v"(lds_off), "v"((uint64_t)(uintptr_t)gptr)
                 : "memory");
}
__device__ __forceinline__ void wait_asynccnt0() {
    asm volatile("s_wait_asynccnt 0x0" ::: "memory");
}

// ---------------------------------------------------------------------------
// Kernel 1: convert both f32 images to f16 in workspace
// ---------------------------------------------------------------------------
__global__ void convert_to_f16(const float* __restrict__ src2,
                               const float* __restrict__ src1,
                               _Float16* __restrict__ dst2,
                               _Float16* __restrict__ dst1, int n) {
    int i = blockIdx.x * blockDim.x + threadIdx.x;
    if (i < n) {
        dst2[i] = (_Float16)src2[i];
        dst1[i] = (_Float16)src1[i];
    }
}

// ---------------------------------------------------------------------------
// Kernel 2: per-patch (mean, 1/std) in f32. One wave32 per patch.
// Each patch row = 5 cols * 64 ch = 320 contiguous floats -> 10 per lane.
// ---------------------------------------------------------------------------
__global__ __launch_bounds__(256) void patch_stats(
    const float* __restrict__ in2, const float* __restrict__ in1,
    float2* __restrict__ stat2, float2* __restrict__ stat1) {
    int wid  = (blockIdx.x * blockDim.x + threadIdx.x) >> 5;
    int lane = threadIdx.x & 31;
    const int NP2 = Bn * M_TOT, NP1 = Bn * N_TOT;
    const float* src; float2* dst; int b, py, px;
    if (wid < NP2) {
        b = wid / M_TOT; int p = wid % M_TOT;
        py = p / P2W; px = p % P2W;
        src = in2; dst = stat2 + wid;
    } else {
        int w1 = wid - NP2;
        if (w1 >= NP1) return;
        b = w1 / N_TOT; int q = w1 % N_TOT;
        py = (q / P1W) * 2; px = (q % P1W) * 2;
        src = in1; dst = stat1 + w1;
    }
    size_t base = ((size_t)(b * Hh + py) * Wd + px) * Cc;
    float s = 0.f, sq = 0.f;
    for (int y = 0; y < PATCH; ++y) {
        const float* row = src + base + (size_t)y * Wd * Cc;
#pragma unroll
        for (int j = 0; j < (PATCH * Cc) / 32; ++j) {  // 10 per lane
            float v = row[j * 32 + lane];
            s += v; sq += v * v;
        }
    }
#pragma unroll
    for (int o = 16; o > 0; o >>= 1) {
        s  += __shfl_down(s, o, 32);
        sq += __shfl_down(sq, o, 32);
    }
    if (lane == 0) {
        float mean = s * (1.f / K_TOT);
        float var  = sq * (1.f / K_TOT) - mean * mean;
        float inv  = rsqrtf(fmaxf(var, 1e-20f));
        *dst = make_float2(mean, inv);
    }
}

// ---------------------------------------------------------------------------
// Kernel 3: double-buffered WMMA GEMM. Tiles staged with
// global_load_async_to_lds_b128 (ASYNCcnt) overlapping the WMMAs; one
// workgroup barrier per K-step. Block = 256 threads = 8 waves (4Mx2N),
// tile = 128(M) x 64(N) x 32(K).
// raw = A(raw p2 patches) x B(raw p1 patches)^T
// out = (raw - K*mu2*mu1) * is2 * is1
// ---------------------------------------------------------------------------
__global__ __launch_bounds__(256) void corr_gemm(
    const _Float16* __restrict__ img2h, const _Float16* __restrict__ img1h,
    const float2* __restrict__ stat2, const float2* __restrict__ stat1,
    float* __restrict__ out) {
    __shared__ _Float16 Alds[2][BM * LDA];
    __shared__ _Float16 Blds[2][BN * LDA];

    const int tid   = threadIdx.x;
    const int b     = blockIdx.z;
    const int tileM = blockIdx.y * BM;
    const int tileN = blockIdx.x * BN;

    // ---- cooperative loader assignment: thread -> (row, 8-half chunk) ------
    const int lrow = tid >> 2;          // 0..63
    const int kc   = (tid & 3) * 8;     // 0,8,16,24 (halves within BK)

    auto origin2 = [&](int m) {         // dense patch origin (input_2)
        int p = m < M_TOT ? m : M_TOT - 1;
        int py = p / P2W, px = p % P2W;
        return (size_t)((b * Hh + py) * Wd + px) * Cc;
    };
    auto origin1 = [&](int n) {         // strided patch origin (input_1)
        int q = n < N_TOT ? n : N_TOT - 1;
        int qy = (q / P1W) * 2, qx = (q % P1W) * 2;
        return (size_t)((b * Hh + qy) * Wd + qx) * Cc;
    };
    const size_t baseA0 = origin2(tileM + lrow);
    const size_t baseA1 = origin2(tileM + lrow + 64);
    const size_t baseB  = origin1(tileN + lrow);

    // per-thread LDS destinations (raw LDS offsets = low 32 bits of flat addr)
    uint32_t ldsA0[2], ldsA1[2], ldsB[2];
#pragma unroll
    for (int s = 0; s < 2; ++s) {
        ldsA0[s] = (uint32_t)(uintptr_t)&Alds[s][lrow * LDA + kc];
        ldsA1[s] = (uint32_t)(uintptr_t)&Alds[s][(lrow + 64) * LDA + kc];
        ldsB[s]  = (uint32_t)(uintptr_t)&Blds[s][lrow * LDA + kc];
    }

    // feature index k -> image offset for this thread's chunk
    auto chunk_off = [&](int kt) {
        int k    = kt + kc;
        int widx = k >> 6;              // dy*5 + dx
        int c    = k & 63;
        int dxx  = widx % 5, dyy = widx / 5;
        return (size_t)((dyy * Wd + dxx) * Cc + c);
    };
    auto issue_tile = [&](int buf, int kt) {
        size_t off = chunk_off(kt);
        async_load_b128(ldsA0[buf], img2h + baseA0 + off);
        async_load_b128(ldsA1[buf], img2h + baseA1 + off);
        async_load_b128(ldsB[buf],  img1h + baseB  + off);
    };

    // ---- wave layout -------------------------------------------------------
    const int wave = tid >> 5, lane = tid & 31;
    const int wm = wave >> 1, wn = wave & 1;  // 4x2 wave grid
    const int lm = lane & 15, lh = lane >> 4;

    v8f acc[2][2];
#pragma unroll
    for (int i = 0; i < 2; ++i)
#pragma unroll
        for (int j = 0; j < 2; ++j) acc[i][j] = {};

    // prologue: stage first tile
    issue_tile(0, 0);
    wait_asynccnt0();
    __syncthreads();

    for (int it = 0; it < NIT; ++it) {
        const int cur = it & 1;
        if (it + 1 < NIT) issue_tile(cur ^ 1, (it + 1) * BK);  // overlap w/ WMMA

        // ---- build fragments per documented wave32 VGPR layouts ------------
        AFrag a[2], bf[2];
#pragma unroll
        for (int mi = 0; mi < 2; ++mi) {
            int arow = wm * 32 + mi * 16 + lm;   // M index for this lane
            // A 16x32 f16: elems 0-7 -> K = lh*8+0..7 ; elems 8-15 -> 16+lh*8+0..7
            a[mi].q[0] = *reinterpret_cast<const uint4*>(&Alds[cur][arow * LDA + lh * 8]);
            a[mi].q[1] = *reinterpret_cast<const uint4*>(&Alds[cur][arow * LDA + 16 + lh * 8]);
        }
#pragma unroll
        for (int ni = 0; ni < 2; ++ni) {
            int brow = wn * 32 + ni * 16 + lm;   // N index for this lane
            // B 32x16 f16: elems 0-7 -> K = lh*8+0..7 ; elems 8-15 -> 16+lh*8+0..7
            bf[ni].q[0] = *reinterpret_cast<const uint4*>(&Blds[cur][brow * LDA + lh * 8]);
            bf[ni].q[1] = *reinterpret_cast<const uint4*>(&Blds[cur][brow * LDA + 16 + lh * 8]);
        }
#pragma unroll
        for (int mi = 0; mi < 2; ++mi)
#pragma unroll
            for (int ni = 0; ni < 2; ++ni)
                acc[mi][ni] = __builtin_amdgcn_wmma_f32_16x16x32_f16(
                    false, a[mi].v, false, bf[ni].v,
                    (short)0, acc[mi][ni], false, false);

        wait_asynccnt0();   // next tile (issued above) landed in LDS
        __syncthreads();    // all waves: next tile visible, cur tile free
    }

    // ---- epilogue: normalization fixup + masked store ----------------------
    float2 s1v[2]; bool nok[2]; int ng[2];
#pragma unroll
    for (int ni = 0; ni < 2; ++ni) {
        ng[ni]  = tileN + wn * 32 + ni * 16 + lm;
        nok[ni] = ng[ni] < N_TOT;
        s1v[ni] = nok[ni] ? stat1[b * N_TOT + ng[ni]] : make_float2(0.f, 0.f);
    }
#pragma unroll
    for (int mi = 0; mi < 2; ++mi) {
#pragma unroll
        for (int v = 0; v < 8; ++v) {
            int m = tileM + wm * 32 + mi * 16 + lh * 8 + v;  // D: VGPR v -> M = v + 8*lh
            if (m >= M_TOT) continue;
            float2 s2 = stat2[b * M_TOT + m];
#pragma unroll
            for (int ni = 0; ni < 2; ++ni) {
                if (!nok[ni]) continue;
                float raw = acc[mi][ni][v];
                float val = (raw - (float)K_TOT * s2.x * s1v[ni].x) * s2.y * s1v[ni].y;
                out[((size_t)b * M_TOT + m) * N_TOT + ng[ni]] = val;
            }
        }
    }
}

// ---------------------------------------------------------------------------
extern "C" void kernel_launch(void* const* d_in, const int* in_sizes, int n_in,
                              void* d_out, int out_size, void* d_ws, size_t ws_size,
                              hipStream_t stream) {
    const float* in1 = (const float*)d_in[0];   // input_1 (strided / B matrix)
    const float* in2 = (const float*)d_in[1];   // input_2 (dense / A matrix)
    float* out = (float*)d_out;

    // workspace layout
    char* ws = (char*)d_ws;
    constexpr size_t IMG_ELEMS = (size_t)Bn * Hh * Wd * Cc;      // 1,048,576
    _Float16* img2h = (_Float16*)ws;                             // 2 MB
    _Float16* img1h = (_Float16*)(ws + IMG_ELEMS * 2);           // 2 MB
    float2* stat2   = (float2*)(ws + IMG_ELEMS * 4);             // 8*1680 float2
    float2* stat1   = (float2*)(ws + IMG_ELEMS * 4 + (size_t)Bn * M_TOT * 8);

    int n = (int)IMG_ELEMS;
    convert_to_f16<<<n / 256, 256, 0, stream>>>(in2, in1, img2h, img1h, n);

    int total_patches = Bn * M_TOT + Bn * N_TOT;                 // 16800
    patch_stats<<<(total_patches + 7) / 8, 256, 0, stream>>>(in2, in1, stat2, stat1);

    dim3 grid((N_TOT + BN - 1) / BN,   // 7
              (M_TOT + BM - 1) / BM,   // 14
              Bn);                     // 8
    corr_gemm<<<grid, 256, 0, stream>>>(img2h, img1h, stat2, stat1, out);
}
